// Seq2SeqRNN_10187662426571
// MI455X (gfx1250) — matrix operand order repaired and checked by reading, not compiled, not run on hardware
//
#include <hip/hip_runtime.h>
#include <hip/hip_bf16.h>

// Seq2Seq GRU + attention + 30K-vocab projection for MI455X (gfx1250, wave32).
// All GEMMs (encoder GRU, decoder GRU, 64x1024x30000 projection) run on
// v_wmma_f32_16x16x32_f16 with f32 accumulation, 4-way N register blocking
// (A fragment reused across 4 WMMAs) and a manual 2-stage software pipeline:
// fragments for k-step i are loaded during k-step i-1, so WMMAs wait on
// loads that are a full k-step old while the next step's loads are in flight.
// proj_w is converted to f16 once (61 MB -> resident in 192 MB L2 across all
// 48 decode steps). dists output (369 MB) streams at HBM rate.

typedef __attribute__((ext_vector_type(16))) _Float16 v16h;
typedef __attribute__((ext_vector_type(8)))  _Float16 v8h;
typedef __attribute__((ext_vector_type(8)))  float    v8f;

#define BB   64      // batch
#define SS   48      // sequence length
#define HH   512     // hidden
#define EE   256     // embed
#define VV   30000   // vocab
#define VP   30016   // vocab padded to multiple of 64 (4 N-tiles per wave)
#define KCAT 768     // E + H  (concatenated GRU input)
#define NG   2048    // GRU gemm output width: [r|z|i_n|h_n]
#define PK   1024    // 2H projection K

__device__ __forceinline__ float sigm(float x) { return 1.0f / (1.0f + __expf(-x)); }

// ---------------------------------------------------------------------------
// 16x64 output strip: 4 adjacent 16x16 tiles of A[M,K] @ W[N,K]^T.
// f16 inputs, f32 accum. A pre-offset to its 16-row tile; W pre-offset to the
// first of 4 consecutive 16-row (output column) tiles. Double-buffered:
// k-step i+1's loads are issued before k-step i's WMMAs.
// A-frag (this code's contiguous k order, not the ISA table's; 16-bit A 16x32): lane = half*16 + m;
//   element e -> K = k0 + half*16 + e.
// B-frag: lane = half*16 + n; element e -> K = k0 + half*16 + e.
// ---------------------------------------------------------------------------
template <int K>
__device__ __forceinline__ void wmma_strip4(const _Float16* __restrict__ A, int lda,
                                            const _Float16* __restrict__ W, int ldb,
                                            int lane, v8f acc[4])
{
    const int r    = lane & 15;
    const int half = lane >> 4;
    const _Float16* arow = A + (size_t)r * lda;
    const _Float16* brow = W + (size_t)r * ldb;

    // Prologue: fragments for k0 = 0.
    v8h  alo = *(const v8h*)(arow + half * 16);
    v8h  ahi = *(const v8h*)(arow + 8 + half * 16);
    v16h b0  = *(const v16h*)(brow + (size_t)0 * 16 * ldb + half * 16);
    v16h b1  = *(const v16h*)(brow + (size_t)1 * 16 * ldb + half * 16);
    v16h b2  = *(const v16h*)(brow + (size_t)2 * 16 * ldb + half * 16);
    v16h b3  = *(const v16h*)(brow + (size_t)3 * 16 * ldb + half * 16);

#pragma unroll
    for (int k0 = 0; k0 < K; k0 += 32) {
        // Issue next k-step's loads before this step's WMMAs.
        v8h  alo_n = alo, ahi_n = ahi;
        v16h b0_n = b0, b1_n = b1, b2_n = b2, b3_n = b3;
        const int kn = k0 + 32;
        if (kn < K) {
            alo_n = *(const v8h*)(arow + kn + half * 16);
            ahi_n = *(const v8h*)(arow + kn + 8 + half * 16);
            b0_n  = *(const v16h*)(brow + (size_t)0 * 16 * ldb + kn + half * 16);
            b1_n  = *(const v16h*)(brow + (size_t)1 * 16 * ldb + kn + half * 16);
            b2_n  = *(const v16h*)(brow + (size_t)2 * 16 * ldb + kn + half * 16);
            b3_n  = *(const v16h*)(brow + (size_t)3 * 16 * ldb + kn + half * 16);
        }
        v16h a;
#pragma unroll
        for (int i = 0; i < 8; ++i) { a[i] = alo[i]; a[i + 8] = ahi[i]; }
        acc[0] = __builtin_amdgcn_wmma_f32_16x16x32_f16(false, a, false, b0, (short)0, acc[0], false, false);
        acc[1] = __builtin_amdgcn_wmma_f32_16x16x32_f16(false, a, false, b1, (short)0, acc[1], false, false);
        acc[2] = __builtin_amdgcn_wmma_f32_16x16x32_f16(false, a, false, b2, (short)0, acc[2], false, false);
        acc[3] = __builtin_amdgcn_wmma_f32_16x16x32_f16(false, a, false, b3, (short)0, acc[3], false, false);
        // Rotate buffers (renamed away by the unroll).
        alo = alo_n; ahi = ahi_n;
        b0 = b0_n; b1 = b1_n; b2 = b2_n; b3 = b3_n;
    }
}

// ---------------------------------------------------------------------------
// GRU gemm: g[64,2048] = A16[64,768] @ Wbig[2048,768]^T.
// 4 mt x 32 nt-groups = 128 waves = 16 blocks; 96 WMMAs per wave.
// ---------------------------------------------------------------------------
__global__ void __launch_bounds__(256)
k_gru_gemm(const _Float16* __restrict__ A, const _Float16* __restrict__ W,
           float* __restrict__ g)
{
    const int wave = blockIdx.x * 8 + (threadIdx.x >> 5);  // 0..127
    const int lane = threadIdx.x & 31;
    const int mt = wave >> 5;    // 0..3
    const int ng = wave & 31;    // nt = ng*4 .. ng*4+3
    v8f acc[4] = {};
    wmma_strip4<KCAT>(A + mt * 16 * KCAT, KCAT,
                      W + (size_t)ng * 64 * KCAT, KCAT, lane, acc);
    const int mbase = mt * 16 + (lane >> 4) * 8;
    const int r = lane & 15;
#pragma unroll
    for (int i = 0; i < 4; ++i) {
        const int n = (ng * 4 + i) * 16 + r;
#pragma unroll
        for (int j = 0; j < 8; ++j)
            g[(size_t)(mbase + j) * NG + n] = acc[i][j];
    }
}

// ---------------------------------------------------------------------------
// Projection: logits[64,30000] = cat16[64,1024] @ proj16[VP,1024]^T + bias.
// 4 mt x 469 nt-groups = 1876 waves; store guard n < VV comes AFTER all
// WMMAs (EXEC all-ones during matrix ops); wave-level exit is uniform.
// ---------------------------------------------------------------------------
__global__ void __launch_bounds__(256)
k_proj(const _Float16* __restrict__ A, const _Float16* __restrict__ W,
       const float* __restrict__ bias, float* __restrict__ out)
{
    const int wave = blockIdx.x * 8 + (threadIdx.x >> 5);
    if (wave >= 4 * (VP / 64)) return;   // uniform across the wave
    const int lane = threadIdx.x & 31;
    const int mt = wave & 3;
    const int ng = wave >> 2;            // 0..468
    v8f acc[4] = {};
    wmma_strip4<PK>(A + mt * 16 * PK, PK,
                    W + (size_t)ng * 64 * PK, PK, lane, acc);
    const int mbase = mt * 16 + (lane >> 4) * 8;
    const int r = lane & 15;
#pragma unroll
    for (int i = 0; i < 4; ++i) {
        const int n = (ng * 4 + i) * 16 + r;
        if (n < VV) {
            const float bn = bias[n];
#pragma unroll
            for (int j = 0; j < 8; ++j)
                out[(size_t)(mbase + j) * VV + n] = acc[i][j] + bn;
        }
    }
}

// ---------------------------------------------------------------------------
// Build Wbig [2048,768] f16:
//  rows 0..1023   : [wih(r,z) | whh(r,z)]
//  rows 1024..1535: [wih(n)   | 0       ]   -> i_n
//  rows 1536..2047: [0        | whh(n)  ]   -> h_n
// ---------------------------------------------------------------------------
__global__ void k_build_wbig(const float* __restrict__ wih,
                             const float* __restrict__ whh,
                             _Float16* __restrict__ W)
{
    int idx = blockIdx.x * blockDim.x + threadIdx.x;
    if (idx >= NG * KCAT) return;
    int row = idx / KCAT, col = idx % KCAT;
    float v = 0.0f;
    if (row < 2 * HH) {
        v = (col < EE) ? wih[(size_t)row * EE + col]
                       : whh[(size_t)row * HH + (col - EE)];
    } else if (row < 3 * HH) {
        v = (col < EE) ? wih[(size_t)row * EE + col] : 0.0f;
    } else {
        int rr = row - HH;  // 1024..1535 of whh
        v = (col >= EE) ? whh[(size_t)rr * HH + (col - EE)] : 0.0f;
    }
    W[idx] = (_Float16)v;
}

// proj_w [VV,PK] f32 -> proj16 [VP,PK] f16, zero-padding rows VV..VP-1.
__global__ void k_build_proj16(const float* __restrict__ src,
                               _Float16* __restrict__ dst)
{
    long i = (long)blockIdx.x * blockDim.x + threadIdx.x;
    if (i >= (long)VP * PK) return;
    long row = i / PK;
    dst[i] = (row < VV) ? (_Float16)src[i] : (_Float16)0.0f;
}

__global__ void k_init(float* __restrict__ h, int* __restrict__ tok)
{
    int i = blockIdx.x * blockDim.x + threadIdx.x;
    if (i < BB * HH) h[i] = 0.0f;
    if (i < BB) tok[i] = 0;
}

// A16[b, 0..255] = embed[row(b)]; A16[b, 256..767] = h[b]
__global__ void k_build_a(const float* __restrict__ embed,
                          const int* __restrict__ input,  // [B,S] or nullptr
                          const int* __restrict__ tok, int s,
                          const float* __restrict__ h,
                          _Float16* __restrict__ A16)
{
    int idx = blockIdx.x * blockDim.x + threadIdx.x;
    if (idx >= BB * KCAT) return;
    int b = idx / KCAT, j = idx % KCAT;
    float v;
    if (j < EE) {
        int row = input ? input[b * SS + s] : tok[b];
        v = embed[(size_t)row * EE + j];
    } else {
        v = h[b * HH + (j - EE)];
    }
    A16[idx] = (_Float16)v;
}

// GRU gates from g[64,2048]; updates h in place; optional enc_states write.
__global__ void k_gate(const float* __restrict__ g, const float* __restrict__ bih,
                       const float* __restrict__ bhh, float* __restrict__ h,
                       float* __restrict__ es, int s)
{
    int idx = blockIdx.x * blockDim.x + threadIdx.x;
    if (idx >= BB * HH) return;
    int b = idx / HH, j = idx % HH;
    const float* gb = g + (size_t)b * NG;
    float r  = sigm(gb[j]          + bih[j]          + bhh[j]);
    float z  = sigm(gb[HH + j]     + bih[HH + j]     + bhh[HH + j]);
    float in = gb[2 * HH + j] + bih[2 * HH + j];
    float hn = gb[3 * HH + j] + bhh[2 * HH + j];
    float nn = tanhf(in + r * hn);
    float hp = h[idx];
    float hnew = (1.0f - z) * nn + z * hp;
    h[idx] = hnew;
    if (es) es[((size_t)b * SS + s) * HH + j] = hnew;
}

// Attention + cat build. One block per batch element.
__global__ void __launch_bounds__(256)
k_attn(const float* __restrict__ es,   // [B,S,H]
       const float* __restrict__ h,    // [B,H]
       _Float16* __restrict__ cat16)   // [B,2H] f16
{
    __shared__ float hbuf[HH];
    __shared__ float sc[SS];
    const int b = blockIdx.x;
    const int t = threadIdx.x;
    for (int i = t; i < HH; i += 256) hbuf[i] = h[b * HH + i];
    __syncthreads();
    const int w = t >> 5, lane = t & 31;
    for (int s = w; s < SS; s += 8) {
        float p = 0.0f;
        const float* row = es + ((size_t)b * SS + s) * HH;
        for (int i = lane; i < HH; i += 32) p += row[i] * hbuf[i];
        for (int off = 16; off; off >>= 1) p += __shfl_down(p, off, 32);
        if (lane == 0) sc[s] = p;
    }
    __syncthreads();
    if (t == 0) {
        float m = sc[0];
        for (int s = 1; s < SS; ++s) m = fmaxf(m, sc[s]);
        float sum = 0.0f;
        for (int s = 0; s < SS; ++s) { float e = __expf(sc[s] - m); sc[s] = e; sum += e; }
        float inv = 1.0f / sum;
        for (int s = 0; s < SS; ++s) sc[s] *= inv;
    }
    __syncthreads();
    for (int i = t; i < HH; i += 256) {
        float c = 0.0f;
        for (int s = 0; s < SS; ++s) c += sc[s] * es[((size_t)b * SS + s) * HH + i];
        cat16[(size_t)b * PK + i]      = (_Float16)hbuf[i];
        cat16[(size_t)b * PK + HH + i] = (_Float16)c;
    }
}

// Per-row max/argmax/logsumexp over V; writes token + predicted(as float).
__global__ void __launch_bounds__(256)
k_rowstats(const float* __restrict__ logits, float* __restrict__ lse,
           int* __restrict__ tok, float* __restrict__ pred)
{
    __shared__ float sm[256];
    __shared__ int   si[256];
    __shared__ float ssum[256];
    const int b = blockIdx.x, t = threadIdx.x;
    const float* row = logits + (size_t)b * VV;
    float m = -1e30f; int mi = 0;
    for (int i = t; i < VV; i += 256) {
        float v = row[i];
        if (v > m) { m = v; mi = i; }
    }
    sm[t] = m; si[t] = mi;
    __syncthreads();
    for (int off = 128; off; off >>= 1) {
        if (t < off) {
            if (sm[t + off] > sm[t] ||
                (sm[t + off] == sm[t] && si[t + off] < si[t])) {
                sm[t] = sm[t + off]; si[t] = si[t + off];
            }
        }
        __syncthreads();
    }
    const float gm = sm[0];
    float s0 = 0.0f;
    for (int i = t; i < VV; i += 256) s0 += __expf(row[i] - gm);
    ssum[t] = s0;
    __syncthreads();
    for (int off = 128; off; off >>= 1) {
        if (t < off) ssum[t] += ssum[t + off];
        __syncthreads();
    }
    if (t == 0) {
        lse[b]  = gm + __logf(ssum[0]);
        tok[b]  = si[0];
        pred[b] = (float)si[0];
    }
}

__global__ void k_logsub(float* __restrict__ logits, const float* __restrict__ lse)
{
    size_t i = (size_t)blockIdx.x * blockDim.x + threadIdx.x;
    if (i < (size_t)BB * VV) logits[i] -= lse[i / VV];
}

// ---------------------------------------------------------------------------
extern "C" void kernel_launch(void* const* d_in, const int* in_sizes, int n_in,
                              void* d_out, int out_size, void* d_ws, size_t ws_size,
                              hipStream_t stream)
{
    const int*   input     = (const int*)  d_in[0];
    const float* enc_embed = (const float*)d_in[1];
    const float* enc_wih   = (const float*)d_in[2];
    const float* enc_whh   = (const float*)d_in[3];
    const float* enc_bih   = (const float*)d_in[4];
    const float* enc_bhh   = (const float*)d_in[5];
    const float* dec_embed = (const float*)d_in[6];
    const float* dec_wih   = (const float*)d_in[7];
    const float* dec_whh   = (const float*)d_in[8];
    const float* dec_bih   = (const float*)d_in[9];
    const float* dec_bhh   = (const float*)d_in[10];
    const float* proj_w    = (const float*)d_in[11];
    const float* proj_b    = (const float*)d_in[12];

    // Workspace carve-out (~75 MB total).
    char* ws = (char*)d_ws;
    size_t off = 0;
    auto carve = [&](size_t bytes) -> void* {
        void* p = ws + off;
        off = (off + bytes + 255) & ~(size_t)255;
        return p;
    };
    _Float16* wbig_enc = (_Float16*)carve((size_t)NG * KCAT * 2);
    _Float16* wbig_dec = (_Float16*)carve((size_t)NG * KCAT * 2);
    _Float16* proj16   = (_Float16*)carve((size_t)VP * PK * 2);
    _Float16* A16      = (_Float16*)carve((size_t)BB * KCAT * 2);
    _Float16* cat16    = (_Float16*)carve((size_t)BB * PK * 2);
    float*    g        = (float*)   carve((size_t)BB * NG * 4);
    float*    h        = (float*)   carve((size_t)BB * HH * 4);
    float*    es       = (float*)   carve((size_t)BB * SS * HH * 4);
    float*    lse      = (float*)   carve((size_t)BB * 4);
    int*      tok      = (int*)     carve((size_t)BB * 4);

    // --- one-time prep: f16 weights ---
    k_build_wbig<<<(NG * KCAT + 255) / 256, 256, 0, stream>>>(enc_wih, enc_whh, wbig_enc);
    k_build_wbig<<<(NG * KCAT + 255) / 256, 256, 0, stream>>>(dec_wih, dec_whh, wbig_dec);
    {
        long n = (long)VP * PK;
        k_build_proj16<<<(int)((n + 255) / 256), 256, 0, stream>>>(proj_w, proj16);
    }
    k_init<<<(BB * HH + 255) / 256, 256, 0, stream>>>(h, tok);

    // --- encoder: 48 sequential GRU steps ---
    for (int s = 0; s < SS; ++s) {
        k_build_a<<<(BB * KCAT + 255) / 256, 256, 0, stream>>>(
            enc_embed, input, nullptr, s, h, A16);
        k_gru_gemm<<<16, 256, 0, stream>>>(A16, wbig_enc, g);
        k_gate<<<(BB * HH + 255) / 256, 256, 0, stream>>>(
            g, enc_bih, enc_bhh, h, es, s);
    }

    // --- decoder: 48 sequential steps with attention + projection ---
    float* out   = (float*)d_out;
    float* dists = out + (size_t)BB * SS;  // predicted [S,B] first, then dists [S,B,V]
    const int proj_waves  = 4 * (VP / 64);            // 1876
    const int proj_blocks = (proj_waves + 7) / 8;     // 235
    for (int s = 0; s < SS; ++s) {
        k_build_a<<<(BB * KCAT + 255) / 256, 256, 0, stream>>>(
            dec_embed, nullptr, tok, s, h, A16);
        k_gru_gemm<<<16, 256, 0, stream>>>(A16, wbig_dec, g);
        k_gate<<<(BB * HH + 255) / 256, 256, 0, stream>>>(
            g, dec_bih, dec_bhh, h, nullptr, 0);
        k_attn<<<BB, 256, 0, stream>>>(es, h, cat16);

        float* lg = dists + (size_t)s * BB * VV;
        k_proj<<<proj_blocks, 256, 0, stream>>>(cat16, proj16, proj_b, lg);
        k_rowstats<<<BB, 256, 0, stream>>>(lg, lse, tok, out + (size_t)s * BB);
        k_logsub<<<((size_t)BB * VV + 255) / 256, 256, 0, stream>>>(lg, lse);
    }
}
